// EnhancedCapsuleLayer_48335561949398
// MI455X (gfx1250) — matrix operand order repaired and checked by reading, not compiled, not run on hardware
//
#include <hip/hip_runtime.h>

// ---------------------------------------------------------------------------
// EnhancedCapsuleLayer for MI455X (gfx1250, wave32)
// B=16, N=2048, D_IN=256, C=32, H=64, HEADS=4, KD=64, ROUTINGS=3
//
//   pack_x / pack_w : fp32 -> bf16 fragment-native layouts (WMMA A/B lane maps)
//   gemm_uhat       : bf16 WMMA GEMM -> u_hat (bf16, 128MB, L2-resident)
//                     B fragments staged to LDS via GLOBAL_LOAD_ASYNC_TO_LDS
//                     epilogue fuses routing iter-0 column sum into s1
//   squash_k        : v = squash(scale * s)
//   route_pass x2   : agreement + softmax(C) + weighted sum, one u_hat read
//   mha_final       : per-batch MHA + residual + LayerNorm + squash * gamma
// ---------------------------------------------------------------------------

typedef __attribute__((ext_vector_type(16))) __bf16 v16bf;
typedef __attribute__((ext_vector_type(8)))  float  v8f;
typedef int v4i __attribute__((vector_size(16)));

#define LDS_AS    __attribute__((address_space(3)))
#define GLOBAL_AS __attribute__((address_space(1)))

__device__ __forceinline__ unsigned short f2bf(float f) {
    unsigned int u = __float_as_uint(f);
    unsigned int r = u + 0x7FFFu + ((u >> 16) & 1u);   // round-to-nearest-even
    return (unsigned short)(r >> 16);
}
__device__ __forceinline__ float bf2f(unsigned short s) {
    return __uint_as_float(((unsigned int)s) << 16);
}

// ---------------- pack x into WMMA A-fragment layout -----------------------
// xpack fragment id = (mt*8 + kt)*32 + lane, 16 bf16 per fragment.
// A 16x32 (16-bit): lane<16 -> kb=0, lane>=16 -> kb=8;
//   elem e<8 : K = kb + e ; elem e>=8 : K = kb + 8 + e   (i.e. kb+16..kb+23)
__global__ void __launch_bounds__(256) pack_x(const float* __restrict__ x,
                                              unsigned short* __restrict__ xpack) {
    int fid  = blockIdx.x * 256 + threadIdx.x;     // 0 .. 2048*8*32-1
    int lane = fid & 31;
    int kt   = (fid >> 5) & 7;
    int mt   = fid >> 8;
    int m    = lane & 15;
    int kb   = (lane >> 4) * 8;
    const float* row = x + (size_t)(mt * 16 + m) * 256 + kt * 32 + kb;
    unsigned short* dst = xpack + (size_t)fid * 16;
#pragma unroll
    for (int e = 0; e < 8; ++e)  dst[e] = f2bf(row[e]);
#pragma unroll
    for (int e = 8; e < 16; ++e) dst[e] = f2bf(row[8 + e]);   // K = kb+16..kb+23
}

// ---------------- pack W into WMMA B-fragment layout -----------------------
// wpack fragment id = ((c*4 + nt)*8 + kt)*32 + lane.
// B 32x16 (16-bit): column n = lane&15, kb = (lane>>4)*8;
//   elem e<8 : K = kb + e ; elem e>=8 : K = kb + 8 + e   (same as A)
__global__ void __launch_bounds__(256) pack_w(const float* __restrict__ W,
                                              unsigned short* __restrict__ wpack) {
    int fid  = blockIdx.x * 256 + threadIdx.x;     // 0 .. 32*4*8*32-1
    int lane = fid & 31;
    int kt   = (fid >> 5) & 7;
    int nt   = (fid >> 8) & 3;
    int c    = fid >> 10;
    int n    = lane & 15;
    int kb   = (lane >> 4) * 8;
    int h    = nt * 16 + n;
    const float* base = W + (size_t)c * 16384 + (size_t)(kt * 32 + kb) * 64 + h;
    unsigned short* dst = wpack + (size_t)fid * 16;
#pragma unroll
    for (int e = 0; e < 16; ++e) dst[e] = f2bf(base[(e < 8 ? e : 8 + e) * 64]);
}

// ---------------- main GEMM: u_hat = x @ W + b_caps ------------------------
// grid (256 mblocks, 32 c), 256 threads = 8 waves; wave -> 16x64 strip.
// The 32KB of B fragments for capsule c are shared by all 8 waves: stage them
// into LDS once via async load-to-LDS, then WMMA streams A from global
// (prefetched) and B from LDS. Epilogue fuses routing iteration 0.
__global__ void __launch_bounds__(256) gemm_uhat(const unsigned short* __restrict__ xpack,
                                                 const unsigned short* __restrict__ wpack,
                                                 const float* __restrict__ b_caps,
                                                 unsigned short* __restrict__ uhat,
                                                 float* __restrict__ s1) {
    __shared__ __align__(32) unsigned short Bsh[8192];   // 16 KB: 8192 bf16 (loop below stages 32 KB)
    __shared__ float ssum[64];
    int t = threadIdx.x;
    if (t < 64) ssum[t] = 0.f;

    int mblock = blockIdx.x;            // 0..255 (128-row super-tile)
    int c      = blockIdx.y;            // 0..31
    int wave   = t >> 5;
    int lane   = t & 31;
    int mt     = mblock * 8 + wave;     // 16-row tile index

    // ---- async stage of B fragments: 2048 x 16B units, 8 per thread ----
    const unsigned short* wsrc = wpack + (size_t)c * 8192;
#pragma unroll
    for (int i = 0; i < 8; ++i) {
        int idx = i * 256 + t;          // 16-byte unit index
#if __has_builtin(__builtin_amdgcn_global_load_async_to_lds_b128)
        GLOBAL_AS v4i* gsrc =
            (GLOBAL_AS v4i*)(unsigned long long)(wsrc + idx * 8);
        LDS_AS v4i* ldst =
            (LDS_AS v4i*)(unsigned int)(unsigned long long)(&Bsh[idx * 8]);
        __builtin_amdgcn_global_load_async_to_lds_b128(gsrc, ldst, 0, 0);
#else
        unsigned int lo = (unsigned int)(unsigned long long)(&Bsh[idx * 8]);
        unsigned long long ga = (unsigned long long)(wsrc + idx * 8);
        asm volatile("global_load_async_to_lds_b128 %0, %1, off"
                     :: "v"(lo), "v"(ga) : "memory");
#endif
    }
#if __has_builtin(__builtin_amdgcn_s_wait_asynccnt)
    __builtin_amdgcn_s_wait_asynccnt(0);
#else
    asm volatile("s_wait_asynccnt 0x0" ::: "memory");
#endif
    __syncthreads();

    const v16bf* apf = ((const v16bf*)xpack) + (size_t)mt * 8 * 32 + lane;
    const v16bf* bsh = (const v16bf*)Bsh;

    v8f acc[4] = {};
#pragma unroll
    for (int kt = 0; kt < 8; ++kt) {
        v16bf a = apf[kt * 32];
        if (kt < 7) __builtin_prefetch((const void*)&apf[(kt + 1) * 32], 0, 1);
#pragma unroll
        for (int nt = 0; nt < 4; ++nt) {
            v16bf b = bsh[(nt * 8 + kt) * 32 + lane];
            acc[nt] = __builtin_amdgcn_wmma_f32_16x16x32_bf16(
                false, a, false, b, (short)0, acc[nt], false, false);
        }
    }

    // C/D layout: VGPR r -> row r (lanes 0-15) / row 8+r (lanes 16-31), col = lane&15
    int ncol    = lane & 15;
    int rowbase = mt * 16 + ((lane >> 4) << 3);
#pragma unroll
    for (int nt = 0; nt < 4; ++nt) {
        int h = nt * 16 + ncol;
        float bias = b_caps[c * 64 + h];
        float csum = 0.f;
#pragma unroll
        for (int r = 0; r < 8; ++r) {
            float val = acc[nt][r] + bias;
            csum += val;
            uhat[(size_t)(rowbase + r) * 2048 + c * 64 + h] = f2bf(val);
        }
        csum += __shfl_xor(csum, 16);              // combine the two row halves
        if (lane < 16) atomicAdd(&ssum[h], csum);  // ds_add_f32
    }
    __syncthreads();
    if (t < 64) {
        int b = mblock >> 4;                       // 16 mblocks per batch
        atomicAdd(&s1[(size_t)(b * 32 + c) * 64 + t], ssum[t]);
    }
}

// ---------------- squash: v = squash(scale * s) over H=64 ------------------
__global__ void __launch_bounds__(256) squash_k(const float* __restrict__ s,
                                                float* __restrict__ v,
                                                float scale, int rows) {
    int wave = threadIdx.x >> 5, lane = threadIdx.x & 31;
    int rowi = blockIdx.x * 8 + wave;
    if (rowi >= rows) return;
    float a0 = s[(size_t)rowi * 64 + lane] * scale;
    float a1 = s[(size_t)rowi * 64 + lane + 32] * scale;
    float ss = a0 * a0 + a1 * a1;
#pragma unroll
    for (int off = 16; off > 0; off >>= 1) ss += __shfl_xor(ss, off);
    float sc = ss / (1.f + ss) * rsqrtf(ss + 1e-7f);
    v[(size_t)rowi * 64 + lane]      = sc * a0;
    v[(size_t)rowi * 64 + lane + 32] = sc * a1;
}

// ---------------- one routing pass -----------------------------------------
// For each (b, n): b_new = b_prev + <u_hat, v>; c = softmax_C(b_new);
// s_out += c * u_hat. One u_hat read per pass (L2-resident bf16).
__global__ void __launch_bounds__(256) route_pass(const unsigned short* __restrict__ uhat,
                                                  const float* __restrict__ vin,
                                                  const float* __restrict__ bprev,
                                                  float* __restrict__ bout,
                                                  float* __restrict__ sout) {
    __shared__ float row[2048];
    __shared__ float vsh[2048];
    __shared__ float apart[256];
    __shared__ float cvec[32];

    int t     = threadIdx.x;
    int chunk = blockIdx.x;          // 0..15 (128 n each)
    int b     = blockIdx.y;          // 0..15
    int cown  = t >> 3;              // capsule owned by this thread
    int pos0  = t * 8;               // 8 contiguous (c,h) slots

    for (int j = 0; j < 8; ++j) vsh[pos0 + j] = vin[(size_t)b * 2048 + pos0 + j];
    float sreg[8] = {0.f, 0.f, 0.f, 0.f, 0.f, 0.f, 0.f, 0.f};
    __syncthreads();

    for (int n = 0; n < 128; ++n) {
        int ng = chunk * 128 + n;
        const unsigned short* up = uhat + (size_t)(b * 2048 + ng) * 2048 + pos0;
        float ag = 0.f;
#pragma unroll
        for (int j = 0; j < 8; ++j) {
            float u = bf2f(up[j]);
            row[pos0 + j] = u;
            ag += u * vsh[pos0 + j];
        }
        apart[t] = ag;
        __syncthreads();
        if (t < 32) {                 // wave 0: reduce + softmax over C
            float a = 0.f;
#pragma unroll
            for (int i = 0; i < 8; ++i) a += apart[t * 8 + i];
            if (bprev) a += bprev[(size_t)(b * 2048 + ng) * 32 + t];
            if (bout)  bout[(size_t)(b * 2048 + ng) * 32 + t] = a;
            float m = a;
#pragma unroll
            for (int off = 16; off > 0; off >>= 1) m = fmaxf(m, __shfl_xor(m, off));
            float e = __expf(a - m);
            float se = e;
#pragma unroll
            for (int off = 16; off > 0; off >>= 1) se += __shfl_xor(se, off);
            cvec[t] = e / se;
        }
        __syncthreads();
        float cv = cvec[cown];
#pragma unroll
        for (int j = 0; j < 8; ++j) sreg[j] += cv * row[pos0 + j];
        __syncthreads();
    }
    for (int j = 0; j < 8; ++j)
        atomicAdd(&sout[(size_t)b * 2048 + pos0 + j], sreg[j]);
}

// ---------------- MHA + residual + LayerNorm + squash*gamma ----------------
__global__ void __launch_bounds__(256) mha_final(const float* __restrict__ routed,
                                                 const float* __restrict__ Wq, const float* __restrict__ bq,
                                                 const float* __restrict__ Wk, const float* __restrict__ bk,
                                                 const float* __restrict__ Wv, const float* __restrict__ bv,
                                                 const float* __restrict__ Wo, const float* __restrict__ bo,
                                                 const float* __restrict__ ln_g, const float* __restrict__ ln_b,
                                                 const float* __restrict__ gamma,
                                                 float* __restrict__ out) {
    __shared__ float X[2048], Qs[2048], Ks[2048], Vs[2048], OUTs[2048], S[1024];
    __shared__ float mu[32], rs[32], rowscale[32];
    int b = blockIdx.x, t = threadIdx.x;

    for (int j = 0; j < 8; ++j) {
        X[t * 8 + j]    = routed[(size_t)b * 2048 + t * 8 + j];
        OUTs[t * 8 + j] = 0.f;
    }
    __syncthreads();

    for (int head = 0; head < 4; ++head) {
        for (int j = 0; j < 8; ++j) {            // q,k,v projections
            int p = t * 8 + j, c = p >> 6, d = p & 63;
            float q = bq[head * 64 + d], k = bk[head * 64 + d], v = bv[head * 64 + d];
            for (int h = 0; h < 64; ++h) {
                float xv = X[c * 64 + h];
                int wi = (h * 4 + head) * 64 + d;
                q += xv * Wq[wi]; k += xv * Wk[wi]; v += xv * Wv[wi];
            }
            Qs[p] = q; Ks[p] = k; Vs[p] = v;
        }
        __syncthreads();
        for (int j = 0; j < 4; ++j) {            // scores / sqrt(KD)
            int p = t * 4 + j, qc = p >> 5, kc = p & 31;
            float s = 0.f;
            for (int d = 0; d < 64; ++d) s += Qs[qc * 64 + d] * Ks[kc * 64 + d];
            S[p] = s * 0.125f;
        }
        __syncthreads();
        if (t < 32) {                            // row softmax
            float m = -1e30f;
            for (int k2 = 0; k2 < 32; ++k2) m = fmaxf(m, S[t * 32 + k2]);
            float sum = 0.f;
            for (int k2 = 0; k2 < 32; ++k2) { float e = __expf(S[t * 32 + k2] - m); S[t * 32 + k2] = e; sum += e; }
            float inv = 1.f / sum;
            for (int k2 = 0; k2 < 32; ++k2) S[t * 32 + k2] *= inv;
        }
        __syncthreads();
        for (int j = 0; j < 8; ++j) {            // ctx -> reuse Qs
            int p = t * 8 + j, qc = p >> 6, d = p & 63;
            float s = 0.f;
            for (int k2 = 0; k2 < 32; ++k2) s += S[qc * 32 + k2] * Vs[k2 * 64 + d];
            Qs[p] = s;
        }
        __syncthreads();
        for (int j = 0; j < 8; ++j) {            // output projection
            int p = t * 8 + j, qc = p >> 6, h = p & 63;
            float s = 0.f;
            for (int d = 0; d < 64; ++d) s += Qs[qc * 64 + d] * Wo[(head * 64 + d) * 64 + h];
            OUTs[p] += s;
        }
        __syncthreads();
    }

    for (int j = 0; j < 8; ++j) { int p = t * 8 + j; OUTs[p] += bo[p & 63] + X[p]; }
    __syncthreads();
    if (t < 32) {                                // LayerNorm stats per capsule
        float m = 0.f;
        for (int h = 0; h < 64; ++h) m += OUTs[t * 64 + h];
        m *= (1.f / 64.f);
        float var = 0.f;
        for (int h = 0; h < 64; ++h) { float d = OUTs[t * 64 + h] - m; var += d * d; }
        mu[t] = m; rs[t] = rsqrtf(var * (1.f / 64.f) + 1e-3f);
    }
    __syncthreads();
    for (int j = 0; j < 8; ++j) {
        int p = t * 8 + j, c = p >> 6, h = p & 63;
        Vs[p] = (OUTs[p] - mu[c]) * rs[c] * ln_g[h] + ln_b[h];
    }
    __syncthreads();
    if (t < 32) {                                // squash scale * gamma
        float ss = 0.f;
        for (int h = 0; h < 64; ++h) { float v = Vs[t * 64 + h]; ss += v * v; }
        rowscale[t] = ss / (1.f + ss) * rsqrtf(ss + 1e-7f) * gamma[0];
    }
    __syncthreads();
    for (int j = 0; j < 8; ++j) {
        int p = t * 8 + j;
        out[(size_t)b * 2048 + p] = Vs[p] * rowscale[p >> 6];
    }
}

// ---------------------------------------------------------------------------
extern "C" void kernel_launch(void* const* d_in, const int* in_sizes, int n_in,
                              void* d_out, int out_size, void* d_ws, size_t ws_size,
                              hipStream_t stream) {
    (void)in_sizes; (void)n_in; (void)out_size; (void)ws_size;
    const float* x      = (const float*)d_in[0];
    const float* W      = (const float*)d_in[1];
    const float* b_caps = (const float*)d_in[2];
    const float* gamma  = (const float*)d_in[3];
    const float* Wq = (const float*)d_in[4];  const float* bq = (const float*)d_in[5];
    const float* Wk = (const float*)d_in[6];  const float* bk = (const float*)d_in[7];
    const float* Wv = (const float*)d_in[8];  const float* bv = (const float*)d_in[9];
    const float* Wo = (const float*)d_in[10]; const float* bo = (const float*)d_in[11];
    const float* lng = (const float*)d_in[12]; const float* lnb = (const float*)d_in[13];
    float* out = (float*)d_out;

    char* ws = (char*)d_ws;
    size_t off = 0;
    auto alloc = [&](size_t bytes) -> void* {
        void* p = ws + off;
        off += (bytes + 255) & ~(size_t)255;
        return p;
    };
    unsigned short* xpack = (unsigned short*)alloc(2048ull * 8 * 32 * 16 * 2); // 16 MB
    unsigned short* wpack = (unsigned short*)alloc(32ull * 4 * 8 * 32 * 16 * 2); // 1 MB
    unsigned short* uhat  = (unsigned short*)alloc(32768ull * 2048 * 2);       // 128 MB (L2-resident)
    float* s1     = (float*)alloc(32768 * 4);
    float* s2     = (float*)alloc(32768 * 4);
    float* s3     = (float*)alloc(32768 * 4);
    float* v1     = (float*)alloc(32768 * 4);
    float* v2     = (float*)alloc(32768 * 4);
    float* routed = (float*)alloc(32768 * 4);
    float* b1     = (float*)alloc(16ull * 2048 * 32 * 4);                      // 4 MB

    // s1,s2,s3 are contiguous (131072 B each, already 256-aligned)
    (void)hipMemsetAsync(s1, 0, 3 * 131072, stream);

    pack_x<<<2048, 256, 0, stream>>>(x, xpack);
    pack_w<<<128, 256, 0, stream>>>(W, wpack);
    gemm_uhat<<<dim3(256, 32), 256, 0, stream>>>(xpack, wpack, b_caps, uhat, s1);
    squash_k<<<64, 256, 0, stream>>>(s1, v1, 1.f / 32.f, 512);                 // iter 0 (uniform c)
    route_pass<<<dim3(16, 16), 256, 0, stream>>>(uhat, v1, nullptr, b1, s2);   // iter 1
    squash_k<<<64, 256, 0, stream>>>(s2, v2, 1.f, 512);
    route_pass<<<dim3(16, 16), 256, 0, stream>>>(uhat, v2, b1, nullptr, s3);   // iter 2
    squash_k<<<64, 256, 0, stream>>>(s3, routed, 1.f, 512);
    mha_final<<<16, 256, 0, stream>>>(routed, Wq, bq, Wk, bk, Wv, bv, Wo, bo,
                                      lng, lnb, gamma, out);
}
